// PhimoeAttention_6975026888744
// MI455X (gfx1250) — matrix operand order repaired and checked by reading, not compiled, not run on hardware
//
#include <hip/hip_runtime.h>
#include <hip/hip_bf16.h>
#include <type_traits>

typedef __attribute__((ext_vector_type(16))) __bf16 v16bf;
typedef __attribute__((ext_vector_type(8)))  float  v8f;

#define S_LEN  2048
#define HID_D  4096
#define NHEAD  32
#define NKVH   8
#define HDIM   128
#define WIN    1024
#define SCALE_F 0.08838834764831845f  // 1/sqrt(128)

struct FragPair { uint4 a, b; };   // 32 bytes == v16bf

// Hardware bf16 converts (v_cvt_pk_bf16_f32 on gfx1250) instead of manual RNE.
__device__ inline unsigned short f2bf(float f) {
  __bf16 h = (__bf16)f;
  return __builtin_bit_cast(unsigned short, h);
}
__device__ inline unsigned int cvt2bf(float a, float b) {
  unsigned int lo = f2bf(a), hi = f2bf(b);
  return lo | (hi << 16);
}
__device__ inline float bf2f(unsigned short h) {
  unsigned int x = ((unsigned int)h) << 16;
  return __builtin_bit_cast(float, x);
}
__device__ inline v8f wmma_bf16(const FragPair& a, const FragPair& b, v8f c) {
  return __builtin_amdgcn_wmma_f32_16x16x32_bf16(
      false, __builtin_bit_cast(v16bf, a),
      false, __builtin_bit_cast(v16bf, b),
      (short)0, c, false, false);
}

// ---------------------------------------------------------------------------
// Tensor Data Mover: 2-D tile load (global -> LDS) with row padding.
// D# bitfields per CDNA5 ISA.
// ---------------------------------------------------------------------------
#if __has_builtin(__builtin_amdgcn_tensor_load_to_lds) && \
    __has_builtin(__builtin_amdgcn_s_wait_tensorcnt)
#define HAS_TDM 1
typedef __attribute__((ext_vector_type(4))) unsigned int v4u;
typedef __attribute__((ext_vector_type(8))) int v8i;
typedef __attribute__((ext_vector_type(4))) int v4i;

__device__ inline void tdm_load_tile_2d(unsigned lds_off, const void* gptr,
                                        unsigned w_elems, unsigned h_rows,
                                        unsigned long long row_stride_elems,
                                        unsigned pad_int_code, unsigned pad_amt_code) {
  unsigned long long ga = (unsigned long long)(size_t)gptr;
  v4u g0;
  g0[0] = 1u;                                         // count=1, user mode
  g0[1] = lds_off;                                    // LDS byte address
  g0[2] = (unsigned)(ga & 0xFFFFFFFFu);               // global_addr lo
  g0[3] = (unsigned)((ga >> 32) & 0x01FFFFFFu) | (2u << 30);  // addr hi | type=2
  v8i g1;
  unsigned dw0 = (1u << 16)                           // data_size = 2 bytes
               | (1u << 20)                           // pad_enable
               | (pad_int_code << 22) | (pad_amt_code << 25);
  g1[0] = (int)dw0;
  g1[1] = (int)((w_elems & 0xFFFFu) << 16);           // tensor_dim0[15:0] @ [63:48]
  g1[2] = (int)(((w_elems >> 16) & 0xFFFFu)           // tensor_dim0[31:16]
              | ((h_rows & 0xFFFFu) << 16));          // tensor_dim1[15:0]
  g1[3] = (int)(((h_rows >> 16) & 0xFFFFu)            // tensor_dim1[31:16]
              | ((w_elems & 0xFFFFu) << 16));         // tile_dim0
  g1[4] = (int)(h_rows & 0xFFFFu);                    // tile_dim1 (tile_dim2 = 0)
  g1[5] = (int)(row_stride_elems & 0xFFFFFFFFull);    // tensor_dim0_stride lo
  g1[6] = (int)((row_stride_elems >> 32) & 0xFFFFull);
  g1[7] = 0;
  v4i z4 = {0, 0, 0, 0};
#if __clang_major__ >= 23
  v8i z8 = {0, 0, 0, 0, 0, 0, 0, 0};
  __builtin_amdgcn_tensor_load_to_lds(g0, g1, z4, z4, z8, 0);
#else
  __builtin_amdgcn_tensor_load_to_lds(g0, g1, z4, z4, 0);
#endif
}
#endif

// ---------------------------------------------------------------------------
// Tiled GEMM: C[M,N] = A[M,K] * W[K,N] + bias. 128x128 block, 8 waves (2x4).
// A fp32 (converted) or bf16 (copied / TDM); W fp32 converted+transposed.
// ---------------------------------------------------------------------------
template<typename AT, bool OUT_BF16>
__global__ __launch_bounds__(256) void gemm_kernel(const AT* __restrict__ A,
    const float* __restrict__ W, const float* __restrict__ bias,
    void* __restrict__ Cout, int M, int N, int K) {
  constexpr int LDA = 40, LDB = 40;           // bf16 elems, 80B rows (16B-aligned)
  __shared__ unsigned short As[128 * LDA];    // [m][k]
  __shared__ unsigned short Bs[128 * LDB];    // transposed: [n][k]
  const int tid  = threadIdx.x;
  const int lane = tid & 31, wave = tid >> 5;
  const int wm = wave & 1, wn = wave >> 1;    // 2x4 wave grid
  const int hi = lane >> 4, nl = lane & 15;
  const int bm0 = blockIdx.y * 128, bn0 = blockIdx.x * 128;

  const v8f vzero = {0.f,0.f,0.f,0.f,0.f,0.f,0.f,0.f};
  v8f acc[4][2];
#pragma unroll
  for (int mi = 0; mi < 4; ++mi)
#pragma unroll
    for (int ni = 0; ni < 2; ++ni) acc[mi][ni] = vzero;

  // B staging coords: each thread owns one n-row of Bs, half the k range.
  const int bn = tid & 127;
  const int bkh = (tid >> 7) << 4;            // 0 or 16

  for (int k0 = 0; k0 < K; k0 += 32) {
    // ---- stage A tile 128x32 -> bf16 LDS ----
    if constexpr (std::is_same<AT, float>::value) {
#pragma unroll
      for (int it = 0; it < 4; ++it) {
        int flat = tid + it * 256;            // float4 units
        int r = flat >> 3, c4 = (flat & 7) << 2;
        float4 v = *(const float4*)(A + (size_t)(bm0 + r) * K + k0 + c4);
        uint2 p; p.x = cvt2bf(v.x, v.y); p.y = cvt2bf(v.z, v.w);
        *(uint2*)&As[r * LDA + c4] = p;
      }
    } else {
#ifdef HAS_TDM
      // bf16 A tile: pure copy -> Tensor Data Mover (wave 0 issues the DMA).
      // 128 rows x 64B, LDS stride 80B: pad_interval=16 dwords (3), pad=4 dwords (3).
      if (wave == 0)
        tdm_load_tile_2d((unsigned)(size_t)&As[0],
                         A + (size_t)bm0 * K + k0, 32, 128,
                         (unsigned long long)K, 3, 3);
#else
#pragma unroll
      for (int it = 0; it < 2; ++it) {
        int flat = tid + it * 256;            // uint4 = 8 bf16
        int r = flat >> 2, c8 = (flat & 3) << 3;
        uint4 v = *(const uint4*)(A + (size_t)(bm0 + r) * K + k0 + c8);
        *(uint4*)&As[r * LDA + c8] = v;
      }
#endif
    }
    // ---- stage W 32x128 -> bf16 transposed LDS [n][k] ----
    {
      float w[16];
      const float* wp = W + (size_t)(k0 + bkh) * N + bn0 + bn;
      __builtin_prefetch(wp + (size_t)32 * N, 0, 1);
#pragma unroll
      for (int j = 0; j < 16; ++j) w[j] = wp[(size_t)j * N];
      uint4 p0, p1;
      p0.x = cvt2bf(w[0], w[1]);  p0.y = cvt2bf(w[2], w[3]);
      p0.z = cvt2bf(w[4], w[5]);  p0.w = cvt2bf(w[6], w[7]);
      p1.x = cvt2bf(w[8], w[9]);  p1.y = cvt2bf(w[10], w[11]);
      p1.z = cvt2bf(w[12], w[13]); p1.w = cvt2bf(w[14], w[15]);
      *(uint4*)&Bs[bn * LDB + bkh]     = p0;
      *(uint4*)&Bs[bn * LDB + bkh + 8] = p1;
    }
#ifdef HAS_TDM
    if constexpr (!std::is_same<AT, float>::value)
      if (wave == 0) __builtin_amdgcn_s_wait_tensorcnt(0);
#endif
    __syncthreads();

    // ---- fragments per ISA 7.12.2 layouts ----
    FragPair af[4], bfr[2];
#pragma unroll
    for (int mi = 0; mi < 4; ++mi) {
      int r = wm * 64 + mi * 16 + nl;
      int o = hi * 8;                         // lanes>=16 hold K=8..15 / 24..31
      af[mi].a = *(const uint4*)&As[r * LDA + o];
      af[mi].b = *(const uint4*)&As[r * LDA + o + 16];
    }
#pragma unroll
    for (int ni = 0; ni < 2; ++ni) {
      int n = wn * 32 + ni * 16 + nl;
      int o = hi * 8;                         // lanes>=16 hold K=8..15 / 24..31
      bfr[ni].a = *(const uint4*)&Bs[n * LDB + o];
      bfr[ni].b = *(const uint4*)&Bs[n * LDB + o + 16];
    }
#pragma unroll
    for (int mi = 0; mi < 4; ++mi)
#pragma unroll
      for (int ni = 0; ni < 2; ++ni)
        acc[mi][ni] = wmma_bf16(af[mi], bfr[ni], acc[mi][ni]);
    __syncthreads();
  }

  // ---- epilogue: bias + store (C layout: VGPR r -> m=r+8*hi, lane%16 -> n) ----
#pragma unroll
  for (int mi = 0; mi < 4; ++mi)
#pragma unroll
    for (int ni = 0; ni < 2; ++ni) {
      int n = bn0 + wn * 32 + ni * 16 + nl;
      float bv = bias[n];
      int rb = bm0 + wm * 64 + mi * 16 + hi * 8;
#pragma unroll
      for (int r = 0; r < 8; ++r) {
        float val = acc[mi][ni][r] + bv;
        if constexpr (OUT_BF16)
          ((unsigned short*)Cout)[(size_t)(rb + r) * N + n] = f2bf(val);
        else
          ((float*)Cout)[(size_t)(rb + r) * N + n] = val;
      }
    }
}

// ---------------------------------------------------------------------------
// RoPE in-place on bf16 [S, nheads, 128]; one thread handles pair (d, d+64).
// ---------------------------------------------------------------------------
__global__ __launch_bounds__(256) void rope_kernel(unsigned short* buf,
    const float* __restrict__ cosT, const float* __restrict__ sinT,
    const int* __restrict__ pos_ids, int nheads) {
  int idx = blockIdx.x * blockDim.x + threadIdx.x;   // S*nheads*64
  int d = idx & 63;
  int h = (idx >> 6) % nheads;
  int s = idx / (64 * nheads);
  int pos = pos_ids[s];
  float c  = cosT[pos * HDIM + d];   // cos[d] == cos[d+64] (concat(freqs,freqs))
  float sn = sinT[pos * HDIM + d];
  size_t base = (size_t)s * nheads * HDIM + (size_t)h * HDIM + d;
  float x1 = bf2f(buf[base]);
  float x2 = bf2f(buf[base + 64]);
  buf[base]      = f2bf(x1 * c - x2 * sn);
  buf[base + 64] = f2bf(x2 * c + x1 * sn);
}

// ---------------------------------------------------------------------------
// Flash attention, sliding window. Grid: (S/128, NHEAD); 8 waves; each wave
// owns 16 query rows. KV in 64-key chunks via LDS; GQA: kv head = h/4.
// K tile staged by the Tensor Data Mover (row pad 256B+16B); V staged
// transposed manually (TDM cannot transpose).
// ---------------------------------------------------------------------------
__global__ __launch_bounds__(256) void flash_kernel(
    const unsigned short* __restrict__ Qb, const unsigned short* __restrict__ Kb,
    const unsigned short* __restrict__ Vb, unsigned short* __restrict__ Ob) {
  constexpr int LDK = 136, LDV = 72, LDP = 72;
  __shared__ unsigned short Ks[64 * LDK];        // [key][d]  (== B^T for Q*K^T)
  __shared__ unsigned short Vt[128 * LDV];       // [d][key]  (== B^T for P*V)
  __shared__ unsigned short Ps[8 * 16 * LDP];    // per-wave P tiles (bf16)
  const int tid  = threadIdx.x;
  const int lane = tid & 31, wave = tid >> 5;
  const int hi = lane >> 4, nl = lane & 15;
  const int qb0 = blockIdx.x * 128;
  const int h   = blockIdx.y;
  const int kvh = h >> 2;

  // Preload Q fragments (16 rows x 128 d = 4 k-steps) straight from global.
  FragPair qf[4];
  {
    int q = qb0 + wave * 16 + nl;
    const unsigned short* qp = Qb + (size_t)q * (NHEAD * HDIM) + (size_t)h * HDIM;
#pragma unroll
    for (int ks = 0; ks < 4; ++ks) {
      int o = ks * 32 + hi * 8;
      qf[ks].a = *(const uint4*)(qp + o);
      qf[ks].b = *(const uint4*)(qp + o + 16);
    }
  }

  const v8f vzero = {0.f,0.f,0.f,0.f,0.f,0.f,0.f,0.f};
  v8f Oacc[8];
#pragma unroll
  for (int dt = 0; dt < 8; ++dt) Oacc[dt] = vzero;
  float mrun[8], lrun[8];
#pragma unroll
  for (int r = 0; r < 8; ++r) { mrun[r] = 0.0f; lrun[r] = 0.0f; }  // max floored at 0

  int cstart = (qb0 >= WIN) ? ((qb0 - (WIN - 1)) & ~63) : 0;
  for (int c0 = cstart; c0 < qb0 + 128; c0 += 64) {
    // ---- stage K chunk [64][128] (TDM) and V chunk transposed [128][64] ----
#ifdef HAS_TDM
    if (wave == 0)
      tdm_load_tile_2d((unsigned)(size_t)&Ks[0],
                       Kb + (size_t)c0 * (NKVH * HDIM) + (size_t)kvh * HDIM,
                       128, 64, (unsigned long long)(NKVH * HDIM), 5, 3);
#else
#pragma unroll
    for (int it = 0; it < 4; ++it) {
      int flat = tid + it * 256;
      int r = flat >> 4, d8 = (flat & 15) << 3;
      size_t g = (size_t)(c0 + r) * (NKVH * HDIM) + (size_t)kvh * HDIM + d8;
      *(uint4*)&Ks[r * LDK + d8] = *(const uint4*)(Kb + g);
    }
#endif
#pragma unroll
    for (int it = 0; it < 4; ++it) {
      int flat = tid + it * 256;               // 1024 uint4 units
      int r = flat >> 4, d8 = (flat & 15) << 3;
      size_t g = (size_t)(c0 + r) * (NKVH * HDIM) + (size_t)kvh * HDIM + d8;
      uint4 vv = *(const uint4*)(Vb + g);
      const unsigned short* pv = (const unsigned short*)&vv;
#pragma unroll
      for (int j = 0; j < 8; ++j) Vt[(d8 + j) * LDV + r] = pv[j];
    }
#ifdef HAS_TDM
    if (wave == 0) __builtin_amdgcn_s_wait_tensorcnt(0);
#endif
    __syncthreads();

    // ---- scores: 4 key tiles x 4 k-steps of WMMA ----
    v8f Sc[4];
#pragma unroll
    for (int ni = 0; ni < 4; ++ni) {
      v8f s = vzero;
#pragma unroll
      for (int ks = 0; ks < 4; ++ks) {
        FragPair kf;
        int kr = ni * 16 + nl;
        int o = ks * 32 + hi * 8;
        kf.a = *(const uint4*)&Ks[kr * LDK + o];
        kf.b = *(const uint4*)&Ks[kr * LDK + o + 16];
        s = wmma_bf16(qf[ks], kf, s);
      }
      Sc[ni] = s;
    }

    // ---- scale + sliding-window causal mask ----
    int qbase = qb0 + wave * 16 + hi * 8;
#pragma unroll
    for (int ni = 0; ni < 4; ++ni)
#pragma unroll
      for (int r = 0; r < 8; ++r) {
        int kj = c0 + ni * 16 + nl;
        int qi = qbase + r;
        float v = Sc[ni][r] * SCALE_F;
        bool ok = (kj <= qi) && (kj > qi - WIN);
        Sc[ni][r] = ok ? v : -1e9f;
      }

    // ---- online softmax: row reductions across 16-lane halves ----
    float t[8];
#pragma unroll
    for (int r = 0; r < 8; ++r)
      t[r] = fmaxf(fmaxf(Sc[0][r], Sc[1][r]), fmaxf(Sc[2][r], Sc[3][r]));
#pragma unroll
    for (int off = 1; off <= 8; off <<= 1)
#pragma unroll
      for (int r = 0; r < 8; ++r) t[r] = fmaxf(t[r], __shfl_xor(t[r], off, 32));
    float alpha[8];
#pragma unroll
    for (int r = 0; r < 8; ++r) {
      float mn = fmaxf(mrun[r], t[r]);
      alpha[r] = __expf(mrun[r] - mn);
      mrun[r] = mn;
    }
    float rs[8];
#pragma unroll
    for (int r = 0; r < 8; ++r) rs[r] = 0.f;
#pragma unroll
    for (int ni = 0; ni < 4; ++ni)
#pragma unroll
      for (int r = 0; r < 8; ++r) {
        float p = __expf(Sc[ni][r] - mrun[r]);
        Sc[ni][r] = p;
        rs[r] += p;
      }
#pragma unroll
    for (int off = 1; off <= 8; off <<= 1)
#pragma unroll
      for (int r = 0; r < 8; ++r) rs[r] += __shfl_xor(rs[r], off, 32);
#pragma unroll
    for (int r = 0; r < 8; ++r) lrun[r] = lrun[r] * alpha[r] + rs[r];
#pragma unroll
    for (int dt = 0; dt < 8; ++dt)
#pragma unroll
      for (int r = 0; r < 8; ++r) Oacc[dt][r] *= alpha[r];

    // ---- P (C layout) -> bf16 LDS -> A-fragment layout ----
    unsigned short* pw = &Ps[wave * 16 * LDP];
#pragma unroll
    for (int ni = 0; ni < 4; ++ni)
#pragma unroll
      for (int r = 0; r < 8; ++r) {
        int m = r + hi * 8;
        pw[m * LDP + ni * 16 + nl] = f2bf(Sc[ni][r]);
      }
    asm volatile("s_wait_dscnt 0" ::: "memory");  // own-wave LDS RAW

    FragPair pf[2];
#pragma unroll
    for (int kst = 0; kst < 2; ++kst) {
      int o = kst * 32 + hi * 8;
      pf[kst].a = *(const uint4*)&pw[nl * LDP + o];
      pf[kst].b = *(const uint4*)&pw[nl * LDP + o + 16];
    }
    // ---- O += P * V ----
#pragma unroll
    for (int dt = 0; dt < 8; ++dt)
#pragma unroll
      for (int kst = 0; kst < 2; ++kst) {
        FragPair vf;
        int dr = dt * 16 + nl;
        int o = kst * 32 + hi * 8;
        vf.a = *(const uint4*)&Vt[dr * LDV + o];
        vf.b = *(const uint4*)&Vt[dr * LDV + o + 16];
        Oacc[dt] = wmma_bf16(pf[kst], vf, Oacc[dt]);
      }
    __syncthreads();
  }

  // ---- normalize and write [S, NH*HD] bf16 ----
#pragma unroll
  for (int r = 0; r < 8; ++r) lrun[r] = 1.0f / lrun[r];
#pragma unroll
  for (int dt = 0; dt < 8; ++dt)
#pragma unroll
    for (int r = 0; r < 8; ++r) {
      int row = qb0 + wave * 16 + r + hi * 8;
      int col = h * HDIM + dt * 16 + nl;
      Ob[(size_t)row * (NHEAD * HDIM) + col] = f2bf(Oacc[dt][r] * lrun[r]);
    }
}

// ---------------------------------------------------------------------------
extern "C" void kernel_launch(void* const* d_in, const int* in_sizes, int n_in,
                              void* d_out, int out_size, void* d_ws, size_t ws_size,
                              hipStream_t stream) {
  (void)in_sizes; (void)n_in; (void)out_size; (void)ws_size;
  const float* hidden = (const float*)d_in[0];
  // d_in[1] = attention_mask (recomputed analytically in-kernel)
  const int*   pos    = (const int*)d_in[2];
  const float* cosT   = (const float*)d_in[3];
  const float* sinT   = (const float*)d_in[4];
  const float* Wq = (const float*)d_in[5];  const float* bq = (const float*)d_in[6];
  const float* Wk = (const float*)d_in[7];  const float* bk = (const float*)d_in[8];
  const float* Wv = (const float*)d_in[9];  const float* bv = (const float*)d_in[10];
  const float* Wo = (const float*)d_in[11]; const float* bo = (const float*)d_in[12];

  unsigned short* Qbuf = (unsigned short*)d_ws;                       // 16 MB
  unsigned short* Kbuf = Qbuf + (size_t)S_LEN * (NHEAD * HDIM);       //  4 MB
  unsigned short* Vbuf = Kbuf + (size_t)S_LEN * (NKVH * HDIM);        //  4 MB
  unsigned short* Abuf = Vbuf + (size_t)S_LEN * (NKVH * HDIM);        // 16 MB

  dim3 blk(256);
  gemm_kernel<float, true><<<dim3((NHEAD*HDIM)/128, S_LEN/128), blk, 0, stream>>>(
      hidden, Wq, bq, Qbuf, S_LEN, NHEAD * HDIM, HID_D);
  gemm_kernel<float, true><<<dim3((NKVH*HDIM)/128, S_LEN/128), blk, 0, stream>>>(
      hidden, Wk, bk, Kbuf, S_LEN, NKVH * HDIM, HID_D);
  gemm_kernel<float, true><<<dim3((NKVH*HDIM)/128, S_LEN/128), blk, 0, stream>>>(
      hidden, Wv, bv, Vbuf, S_LEN, NKVH * HDIM, HID_D);
  rope_kernel<<<(S_LEN * NHEAD * 64) / 256, blk, 0, stream>>>(Qbuf, cosT, sinT, pos, NHEAD);
  rope_kernel<<<(S_LEN * NKVH  * 64) / 256, blk, 0, stream>>>(Kbuf, cosT, sinT, pos, NKVH);
  flash_kernel<<<dim3(S_LEN / 128, NHEAD), blk, 0, stream>>>(Qbuf, Kbuf, Vbuf, Abuf);
  gemm_kernel<unsigned short, false><<<dim3(HID_D/128, S_LEN/128), blk, 0, stream>>>(
      Abuf, Wo, bo, d_out, S_LEN, HID_D, NHEAD * HDIM);
}
